// Attention_21569325760724
// MI455X (gfx1250) — matrix operand order repaired and checked by reading, not compiled, not run on hardware
//
#include <hip/hip_runtime.h>

// ---------------------------------------------------------------------------
// MI455X (gfx1250) causal attention: bf16 WMMA pipeline.
//   B=2, S=2048, D_MODEL=1024, H=16, Dh=64, TOKENS=4096
// ---------------------------------------------------------------------------

typedef __bf16 bf16;
typedef bf16  bf16x8  __attribute__((ext_vector_type(8)));
typedef bf16  bf16x16 __attribute__((ext_vector_type(16)));
typedef float floatx8 __attribute__((ext_vector_type(8)));

#define S_LEN   2048
#define DMODEL  1024
#define NHEADS  16
#define DHEAD   64
#define NTOK    4096

__device__ __forceinline__ bf16 f2bf(float f) {
  unsigned u = __builtin_bit_cast(unsigned, f);
  u += 0x7fffu + ((u >> 16) & 1u);          // round-to-nearest-even
  unsigned short h = (unsigned short)(u >> 16);
  return __builtin_bit_cast(bf16, h);
}

__device__ __forceinline__ floatx8 wmma_bf16(bf16x16 a, bf16x16 b, floatx8 c) {
  // D(16x16 f32) = A(16x32 bf16) * B(32x16 bf16) + C
  return __builtin_amdgcn_wmma_f32_16x16x32_bf16(false, a, false, b,
                                                 (short)0, c, false, false);
}

__device__ __forceinline__ bf16x16 cat8(bf16x8 lo, bf16x8 hi) {
  return __builtin_shufflevector(lo, hi, 0,1,2,3,4,5,6,7,8,9,10,11,12,13,14,15);
}

__device__ __forceinline__ void wave_lds_fence() {
#if __has_builtin(__builtin_amdgcn_s_wait_dscnt)
  __builtin_amdgcn_s_wait_dscnt(0);
#else
  asm volatile("s_wait_dscnt 0" ::: "memory");
#endif
}

// ---------------------------------------------------------------------------
// Kernel 0: fp32 -> bf16 conversion (layout preserved)
// ---------------------------------------------------------------------------
__global__ void cvt_f32_to_bf16_kernel(const float* __restrict__ src,
                                       bf16* __restrict__ dst, int n) {
  int i = blockIdx.x * blockDim.x + threadIdx.x;
  if (i < n) dst[i] = f2bf(src[i]);
}

// ---------------------------------------------------------------------------
// Kernel 1: fused QKV projection (WMMA) + RoPE epilogue.
// One wave per 16x16 output tile.  A = x_bf16 [4096][1024] row-major,
// B = W viewed as [n=h*64+d][m] row-major (column-of-B == contiguous row).
// mat 0 -> Q (RoPE, store [b][h][s][d])
// mat 1 -> K (RoPE, store [b][h][s][d])
// mat 2 -> V (store transposed [b][h][d][s])
// ---------------------------------------------------------------------------
__global__ void qkv_rope_wmma_kernel(
    const bf16* __restrict__ xb,
    const bf16* __restrict__ wq, const bf16* __restrict__ wk,
    const bf16* __restrict__ wv,
    const float* __restrict__ sines, const float* __restrict__ cosines,
    bf16* __restrict__ Qb, bf16* __restrict__ Kb, bf16* __restrict__ Vt)
{
  const int lane = threadIdx.x & 31;
  const int wid  = threadIdx.x >> 5;
  const int tile = blockIdx.x * 8 + wid;   // 0 .. 3*16384-1
  const int mat  = tile >> 14;             // 0=Q 1=K 2=V
  const int rem  = tile & 16383;
  const int tm   = rem >> 6;               // token tile  0..255
  const int tn   = rem & 63;               // column tile 0..63
  const int col  = lane & 15;
  const int hi   = lane >> 4;

  const bf16* W = (mat == 0) ? wq : (mat == 1) ? wk : wv;
  const bf16* aptr = xb + (size_t)(tm * 16 + col) * DMODEL;
  const bf16* bptr = W  + (size_t)(tn * 16 + col) * DMODEL;

  floatx8 acc = {0.f,0.f,0.f,0.f,0.f,0.f,0.f,0.f};
  #pragma unroll 4
  for (int kb = 0; kb < DMODEL; kb += 32) {
    bf16x8  a0  = *(const bf16x8*)(aptr + kb + 16 * hi);
    bf16x8  a1  = *(const bf16x8*)(aptr + kb + 16 * hi + 8);
    bf16x16 bfr = *(const bf16x16*)(bptr + kb + 16 * hi);
    acc = wmma_bf16(cat8(a0, a1), bfr, acc);
  }

  const int n = tn * 16 + col;     // h*64 + d
  const int h = n >> 6;
  const int d = n & 63;

  if (mat < 2) {
    const int d2  = d >> 1;
    const int odd = d & 1;
    bf16* dst = (mat == 0) ? Qb : Kb;
    #pragma unroll
    for (int r = 0; r < 8; ++r) {
      const int token = tm * 16 + r + 8 * hi;
      const int p     = token & (S_LEN - 1);
      const int bidx  = token >> 11;
      const float cs = cosines[p * 32 + d2];
      const float sn = sines[p * 32 + d2];
      const float v  = acc[r];
      const float pv = __shfl_xor(v, 1, 32);     // RoPE partner column
      const float rv = odd ? (pv * sn + v * cs) : (v * cs - pv * sn);
      dst[(((size_t)bidx * NHEADS + h) * S_LEN + p) * DHEAD + d] = f2bf(rv);
    }
  } else {
    #pragma unroll
    for (int r = 0; r < 8; ++r) {
      const int token = tm * 16 + r + 8 * hi;
      const int p     = token & (S_LEN - 1);
      const int bidx  = token >> 11;
      Vt[(((size_t)bidx * NHEADS + h) * DHEAD + d) * S_LEN + p] = f2bf(acc[r]);
    }
  }
}

// ---------------------------------------------------------------------------
// Kernel 2: flash attention (online softmax, causal), one wave per 16-query
// tile of one (b,h).  32-key blocks: 4 WMMA (Q*K^T) + 4 WMMA (P*V) per block.
// P is transposed C-layout -> A-layout through a private LDS tile, fenced
// with s_wait_dscnt (per-wave, no workgroup barrier).
// ---------------------------------------------------------------------------
__global__ void flash_attn_wmma_kernel(
    const bf16* __restrict__ Qb, const bf16* __restrict__ Kb,
    const bf16* __restrict__ Vt, bf16* __restrict__ AO)
{
  const int lane = threadIdx.x & 31;
  const int wid  = threadIdx.x >> 5;
  const int gw   = blockIdx.x * 8 + wid;   // 0..4095
  const int qb   = gw & 127;               // query tile
  const int bh   = gw >> 7;                // b*16 + h
  const int col  = lane & 15;
  const int hi   = lane >> 4;

  __shared__ __align__(32) bf16 lds[8][16 * 32];
  bf16* pt = lds[wid];

  // Q A-fragments (16 queries x 64 d, two k-steps of 32)
  const bf16* qrow = Qb + ((size_t)bh * S_LEN + qb * 16 + col) * DHEAD;
  bf16x16 qa0 = cat8(*(const bf16x8*)(qrow + 16 * hi),
                     *(const bf16x8*)(qrow + 16 * hi + 8));
  bf16x16 qa1 = cat8(*(const bf16x8*)(qrow + 32 + 16 * hi),
                     *(const bf16x8*)(qrow + 32 + 16 * hi + 8));

  const floatx8 zero = {0.f,0.f,0.f,0.f,0.f,0.f,0.f,0.f};
  floatx8 o0 = zero, o1 = zero, o2 = zero, o3 = zero;
  float mrow[8], lrow[8];
  #pragma unroll
  for (int r = 0; r < 8; ++r) { mrow[r] = -1e30f; lrow[r] = 0.f; }

  const bf16* Kbase = Kb + (size_t)bh * S_LEN * DHEAD;
  const bf16* Vbase = Vt + (size_t)bh * DHEAD * S_LEN;

  const int nkb = (qb * 16 + 47) >> 5;     // causal: key blocks needed
  for (int jb = 0; jb < nkb; ++jb) {
    const int kb0 = jb * 32;

    // --- scores: S[16q x 32k] as two 16x16 C-tiles ---
    floatx8 s0 = zero, s1 = zero;
    {
      const bf16* kr0 = Kbase + (size_t)(kb0 + col) * DHEAD;
      s0 = wmma_bf16(qa0, *(const bf16x16*)(kr0 + 16 * hi), s0);
      s0 = wmma_bf16(qa1, *(const bf16x16*)(kr0 + 32 + 16 * hi), s0);
      const bf16* kr1 = Kbase + (size_t)(kb0 + 16 + col) * DHEAD;
      s1 = wmma_bf16(qa0, *(const bf16x16*)(kr1 + 16 * hi), s1);
      s1 = wmma_bf16(qa1, *(const bf16x16*)(kr1 + 32 + 16 * hi), s1);
    }

    // --- causal mask + online softmax update ---
    #pragma unroll
    for (int r = 0; r < 8; ++r) {
      const int q = qb * 16 + r + 8 * hi;
      float v0 = (kb0 + col      <= q) ? s0[r] : -1e30f;
      float v1 = (kb0 + 16 + col <= q) ? s1[r] : -1e30f;
      float rmx = fmaxf(v0, v1);
      rmx = fmaxf(rmx, __shfl_xor(rmx, 1, 32));
      rmx = fmaxf(rmx, __shfl_xor(rmx, 2, 32));
      rmx = fmaxf(rmx, __shfl_xor(rmx, 4, 32));
      rmx = fmaxf(rmx, __shfl_xor(rmx, 8, 32));
      const float mnew  = fmaxf(mrow[r], rmx);
      const float alpha = __expf(mrow[r] - mnew);
      mrow[r] = mnew;
      const float p0 = __expf(v0 - mnew);
      const float p1 = __expf(v1 - mnew);
      float rs = p0 + p1;
      rs += __shfl_xor(rs, 1, 32);
      rs += __shfl_xor(rs, 2, 32);
      rs += __shfl_xor(rs, 4, 32);
      rs += __shfl_xor(rs, 8, 32);
      lrow[r] = lrow[r] * alpha + rs;
      o0[r] *= alpha; o1[r] *= alpha; o2[r] *= alpha; o3[r] *= alpha;
      // stash P (C-layout) into LDS [q][key] tile
      pt[(r + 8 * hi) * 32 + col]      = f2bf(p0);
      pt[(r + 8 * hi) * 32 + 16 + col] = f2bf(p1);
    }
    wave_lds_fence();

    // --- reload P as A-fragment (row=q, contiguous key) ---
    const bf16* pr = pt + col * 32;
    bf16x16 pa = cat8(*(const bf16x8*)(pr + 16 * hi),
                      *(const bf16x8*)(pr + 16 * hi + 8));

    // --- O += P * V  (V stored [d][s] so B-fragments are contiguous) ---
    const bf16* vb = Vbase + (size_t)col * S_LEN + kb0 + 16 * hi;
    o0 = wmma_bf16(pa, *(const bf16x16*)(vb),               o0);
    o1 = wmma_bf16(pa, *(const bf16x16*)(vb + 16 * S_LEN),  o1);
    o2 = wmma_bf16(pa, *(const bf16x16*)(vb + 32 * S_LEN),  o2);
    o3 = wmma_bf16(pa, *(const bf16x16*)(vb + 48 * S_LEN),  o3);
  }

  // --- epilogue: normalize, store attn output as bf16 [token][h*64+d] ---
  const int bidx = bh >> 4;
  const int h    = bh & 15;
  #pragma unroll
  for (int r = 0; r < 8; ++r) {
    const float inv = 1.0f / lrow[r];
    const size_t tok = (size_t)bidx * S_LEN + qb * 16 + r + 8 * hi;
    bf16* dst = AO + tok * DMODEL + h * DHEAD + col;
    dst[0]  = f2bf(o0[r] * inv);
    dst[16] = f2bf(o1[r] * inv);
    dst[32] = f2bf(o2[r] * inv);
    dst[48] = f2bf(o3[r] * inv);
  }
}

// ---------------------------------------------------------------------------
// Kernel 3: output projection.  A = AO bf16 [4096][1024],
// B = output_weight viewed as [m][h*64+d] (natural layout), C -> fp32 d_out.
// ---------------------------------------------------------------------------
__global__ void outproj_wmma_kernel(const bf16* __restrict__ AO,
                                    const bf16* __restrict__ wo,
                                    float* __restrict__ out)
{
  const int lane = threadIdx.x & 31;
  const int wid  = threadIdx.x >> 5;
  const int tile = blockIdx.x * 8 + wid;   // 0..16383
  const int tm = tile >> 6;
  const int tn = tile & 63;
  const int col = lane & 15;
  const int hi  = lane >> 4;

  const bf16* aptr = AO + (size_t)(tm * 16 + col) * DMODEL;
  const bf16* bptr = wo + (size_t)(tn * 16 + col) * DMODEL;

  floatx8 acc = {0.f,0.f,0.f,0.f,0.f,0.f,0.f,0.f};
  #pragma unroll 4
  for (int kb = 0; kb < DMODEL; kb += 32) {
    bf16x8  a0  = *(const bf16x8*)(aptr + kb + 16 * hi);
    bf16x8  a1  = *(const bf16x8*)(aptr + kb + 16 * hi + 8);
    bf16x16 bfr = *(const bf16x16*)(bptr + kb + 16 * hi);
    acc = wmma_bf16(cat8(a0, a1), bfr, acc);
  }
  #pragma unroll
  for (int r = 0; r < 8; ++r) {
    const int token = tm * 16 + r + 8 * hi;
    out[(size_t)token * DMODEL + tn * 16 + col] = acc[r];
  }
}

// ---------------------------------------------------------------------------
extern "C" void kernel_launch(void* const* d_in, const int* in_sizes, int n_in,
                              void* d_out, int out_size, void* d_ws, size_t ws_size,
                              hipStream_t stream) {
  const float* x  = (const float*)d_in[0];
  const float* kw = (const float*)d_in[1];
  const float* qw = (const float*)d_in[2];
  const float* vw = (const float*)d_in[3];
  const float* ow = (const float*)d_in[4];
  const float* sn = (const float*)d_in[5];
  const float* cs = (const float*)d_in[6];

  char* ws = (char*)d_ws;
  const size_t MiB = 1ull << 20;
  bf16* xb = (bf16*)(ws + 0 * MiB);   // 8 MiB  x in bf16
  bf16* wq = (bf16*)(ws + 8 * MiB);   // 2 MiB
  bf16* wk = (bf16*)(ws + 10 * MiB);  // 2 MiB
  bf16* wv = (bf16*)(ws + 12 * MiB);  // 2 MiB
  bf16* wo = (bf16*)(ws + 14 * MiB);  // 2 MiB
  bf16* Qb = (bf16*)(ws + 16 * MiB);  // 8 MiB [b][h][s][d]
  bf16* Kb = (bf16*)(ws + 24 * MiB);  // 8 MiB [b][h][s][d]
  bf16* Vt = (bf16*)(ws + 32 * MiB);  // 8 MiB [b][h][d][s]
  bf16* AO = (bf16*)(ws + 40 * MiB);  // 8 MiB [token][h*64+d]

  cvt_f32_to_bf16_kernel<<<(NTOK * DMODEL) / 256, 256, 0, stream>>>(x, xb, NTOK * DMODEL);
  cvt_f32_to_bf16_kernel<<<(DMODEL * DMODEL) / 256, 256, 0, stream>>>(qw, wq, DMODEL * DMODEL);
  cvt_f32_to_bf16_kernel<<<(DMODEL * DMODEL) / 256, 256, 0, stream>>>(kw, wk, DMODEL * DMODEL);
  cvt_f32_to_bf16_kernel<<<(DMODEL * DMODEL) / 256, 256, 0, stream>>>(vw, wv, DMODEL * DMODEL);
  cvt_f32_to_bf16_kernel<<<(DMODEL * DMODEL) / 256, 256, 0, stream>>>(ow, wo, DMODEL * DMODEL);

  // 3 matrices * 256x64 tiles = 49152 waves, 8 waves/block
  qkv_rope_wmma_kernel<<<6144, 256, 0, stream>>>(xb, wq, wk, wv, sn, cs, Qb, Kb, Vt);
  // 2*16*128 = 4096 waves, 8 waves/block
  flash_attn_wmma_kernel<<<512, 256, 0, stream>>>(Qb, Kb, Vt, AO);
  // 16384 tiles, 8 waves/block
  outproj_wmma_kernel<<<2048, 256, 0, stream>>>(AO, wo, (float*)d_out);
}
